// GPTJBlock_47897475285588
// MI455X (gfx1250) — compile-verified, matrix operand order checked by reading, not run on hardware
//
#include <hip/hip_runtime.h>
#include <hip/hip_bf16.h>

typedef __bf16 bf16;
typedef __bf16 v16bf __attribute__((ext_vector_type(16)));
typedef float  v8f   __attribute__((ext_vector_type(8)));

constexpr int Bc = 2, Sc = 2048, Dc = 4096, Hc = 16, HDc = 256, ROTc = 64;
constexpr int INNERc = 16384, TDc = 3 * Dc;       // 12288
constexpr int MRc = Bc * Sc;                      // 4096 token rows

static __device__ __forceinline__ v8f wmma_bf16(v16bf a, v16bf b, v8f c) {
  return __builtin_amdgcn_wmma_f32_16x16x32_bf16(false, a, false, b, (short)0, c,
                                                 false, false);
}

// ---------------- async global->LDS staging (gfx1250), guarded by probe ----
#if __has_builtin(__builtin_amdgcn_global_load_async_to_lds_b128)
#define HAVE_ASYNC_LDS 1
typedef int v4ig __attribute__((vector_size(4 * sizeof(int))));
typedef __attribute__((address_space(1))) v4ig gv4;   // global int4
typedef __attribute__((address_space(3))) v4ig lv4;   // LDS int4
static __device__ __forceinline__ void async_cp16(void* lds, const void* g) {
  __builtin_amdgcn_global_load_async_to_lds_b128((gv4*)g, (lv4*)lds, 0, 0);
}
static __device__ __forceinline__ void wait_async0() {
#if __has_builtin(__builtin_amdgcn_s_wait_asynccnt)
  __builtin_amdgcn_s_wait_asynccnt(0);
#else
  asm volatile("s_wait_asynccnt 0x0" ::: "memory");
#endif
}
#endif

// ---------------------------------------------------------------- LayerNorm
__global__ __launch_bounds__(256) void ln_kernel(const float* __restrict__ x,
                                                 const float* __restrict__ g,
                                                 const float* __restrict__ bb,
                                                 bf16* __restrict__ h) {
  const int row = blockIdx.x;
  const int tid = threadIdx.x;
  const float* xr = x + (size_t)row * Dc;
  __shared__ float red[256];
  float s = 0.f;
  for (int i = tid; i < Dc; i += 256) s += xr[i];
  red[tid] = s;
  __syncthreads();
  for (int o = 128; o > 0; o >>= 1) {
    if (tid < o) red[tid] += red[tid + o];
    __syncthreads();
  }
  const float mu = red[0] * (1.f / Dc);
  __syncthreads();
  float v = 0.f;
  for (int i = tid; i < Dc; i += 256) {
    float d = xr[i] - mu;
    v += d * d;
  }
  red[tid] = v;
  __syncthreads();
  for (int o = 128; o > 0; o >>= 1) {
    if (tid < o) red[tid] += red[tid + o];
    __syncthreads();
  }
  const float rstd = rsqrtf(red[0] * (1.f / Dc) + 1e-5f);
  bf16* hr = h + (size_t)row * Dc;
  for (int i = tid; i < Dc; i += 256)
    hr[i] = (bf16)((xr[i] - mu) * rstd * g[i] + bb[i]);
}

// ------------------------------------------- fp32 [K][N] -> bf16 [N][K] cast
__global__ void transpose_cast(const float* __restrict__ w, bf16* __restrict__ wT,
                               int K, int N) {
  __shared__ float tile[32][33];
  const int nx = blockIdx.x * 32, kx = blockIdx.y * 32;
  const int tx = threadIdx.x, ty = threadIdx.y;
#pragma unroll
  for (int j = 0; j < 32; j += 8)
    tile[ty + j][tx] = w[(size_t)(kx + ty + j) * N + nx + tx];
  __syncthreads();
#pragma unroll
  for (int j = 0; j < 32; j += 8)
    wT[(size_t)(nx + ty + j) * K + kx + tx] = (bf16)tile[tx][ty + j];
}

// --------------------------------------------------------- WMMA bf16 GEMM
// C[M,N] = A[M,K] * Bt[N,K]^T, 128x128x32 tiles, 8 waves (2x4), 4x2 frags/wave.
// EPI: 0 = store f32 | 1 = +bias+residual f32 | 2 = accumulate f32 | 3 = bias+gelu bf16
template <int EPI>
__global__ __launch_bounds__(256) void gemm_wmma(
    const bf16* __restrict__ A, const bf16* __restrict__ Bt,
    float* __restrict__ outf, bf16* __restrict__ outh,
    const float* __restrict__ bias, const float* __restrict__ resid,
    int M, int N, int K) {
  __shared__ uint4 As4[2][512];  // 128 rows x 32 bf16 per buffer
  __shared__ uint4 Bs4[2][512];
  const int tid = threadIdx.x;
  const int lane = tid & 31;
  const int ln = lane & 15, hl = lane >> 4;
  const int wv = tid >> 5;
  const int mw = (wv >> 2) * 64;   // 0 / 64
  const int nw = (wv & 3) * 32;    // 0 / 32 / 64 / 96
  const int m0 = blockIdx.y * 128;
  const int n0 = blockIdx.x * 128;

  const v8f vzero = {};
  v8f acc[4][2];
#pragma unroll
  for (int i = 0; i < 4; i++)
#pragma unroll
    for (int j = 0; j < 2; j++) acc[i][j] = vzero;

  const uint4* Ag = (const uint4*)(A + (size_t)m0 * K);
  const uint4* Bg = (const uint4*)(Bt + (size_t)n0 * K);
  const bf16* Asb = (const bf16*)As4;
  const bf16* Bsb = (const bf16*)Bs4;
  const int c0 = tid * 2;
  const int nk = K >> 5;

  auto load_tile = [&](int buf, int t) {
    const int kt = t << 5;
#pragma unroll
    for (int j = 0; j < 2; j++) {
      const int ch = c0 + j;
      const int r = ch >> 2, cc = ch & 3;
#ifdef HAVE_ASYNC_LDS
      async_cp16(&As4[buf][ch], &Ag[(((size_t)r * K + kt) >> 3) + cc]);
      async_cp16(&Bs4[buf][ch], &Bg[(((size_t)r * K + kt) >> 3) + cc]);
#else
      As4[buf][ch] = Ag[(((size_t)r * K + kt) >> 3) + cc];
      Bs4[buf][ch] = Bg[(((size_t)r * K + kt) >> 3) + cc];
#endif
    }
    if (t + 2 < nk) {  // speculative prefetch of tile t+2
      const int kt2 = (t + 2) << 5;
      const int r = c0 >> 2;
      __builtin_prefetch((const void*)&Ag[((size_t)r * K + kt2) >> 3], 0, 0);
      __builtin_prefetch((const void*)&Bg[((size_t)r * K + kt2) >> 3], 0, 0);
    }
  };

  auto compute = [&](int buf) {
    v16bf a[4], b[2];
#pragma unroll
    for (int mt = 0; mt < 4; mt++)
      a[mt] = *(const v16bf*)(Asb + buf * 4096 + (mw + mt * 16 + ln) * 32 + hl * 16);
#pragma unroll
    for (int nt = 0; nt < 2; nt++)
      b[nt] = *(const v16bf*)(Bsb + buf * 4096 + (nw + nt * 16 + ln) * 32 + hl * 16);
#pragma unroll
    for (int mt = 0; mt < 4; mt++)
#pragma unroll
      for (int nt = 0; nt < 2; nt++) acc[mt][nt] = wmma_bf16(a[mt], b[nt], acc[mt][nt]);
  };

  load_tile(0, 0);
#ifdef HAVE_ASYNC_LDS
  wait_async0();
#endif
  __syncthreads();
  int buf = 0;
  for (int t = 0; t < nk; t++) {
    if (t + 1 < nk) load_tile(buf ^ 1, t + 1);
    compute(buf);
#ifdef HAVE_ASYNC_LDS
    wait_async0();
#endif
    __syncthreads();
    buf ^= 1;
  }

#pragma unroll
  for (int mt = 0; mt < 4; mt++)
#pragma unroll
    for (int nt = 0; nt < 2; nt++)
#pragma unroll
      for (int i = 0; i < 8; i++) {
        const int row = m0 + mw + mt * 16 + i + hl * 8;
        const int col = n0 + nw + nt * 16 + ln;
        const size_t idx = (size_t)row * N + col;
        const float v = acc[mt][nt][i];
        if constexpr (EPI == 0) {
          outf[idx] = v;
        } else if constexpr (EPI == 1) {
          outf[idx] = v + bias[col] + resid[idx];
        } else if constexpr (EPI == 2) {
          outf[idx] += v;
        } else {
          const float x = v + bias[col];
          const float gx =
              0.5f * x * (1.f + tanhf(0.7978845608028654f * (x + 0.044715f * x * x * x)));
          outh[idx] = (bf16)gx;
        }
      }
}

// ------------------------------------------------------------------- RoPE
__global__ __launch_bounds__(128) void rope_kernel(const float* __restrict__ qkv,
                                                   const int* __restrict__ pos,
                                                   bf16* __restrict__ Qb,
                                                   bf16* __restrict__ Kb,
                                                   bf16* __restrict__ Vt) {
  const int gid = blockIdx.x;          // b*S*H + s*H + h
  const int h = gid % Hc;
  const int s = (gid / Hc) % Sc;
  const int b = gid / (Hc * Sc);
  const int d2 = threadIdx.x;          // 0..127 -> element pair
  const size_t rowbase = (size_t)(b * Sc + s) * TDc;
  const float* qp = qkv + rowbase + h * HDc;
  const float* kp = qp + Dc;
  const float* vp = qp + 2 * Dc;
  float q0 = qp[2 * d2], q1 = qp[2 * d2 + 1];
  float k0 = kp[2 * d2], k1 = kp[2 * d2 + 1];
  if (d2 < ROTc / 2) {
    const float p = (float)pos[s];
    const float inv = __expf(-(2.f * d2 / (float)ROTc) * __logf(10000.f));
    float sn, cs;
    __sincosf(p * inv, &sn, &cs);
    float t;
    t = q0 * cs - q1 * sn; q1 = q1 * cs + q0 * sn; q0 = t;
    t = k0 * cs - k1 * sn; k1 = k1 * cs + k0 * sn; k0 = t;
  }
  const size_t obase = ((size_t)(b * Hc + h) * Sc + s) * HDc;
  Qb[obase + 2 * d2] = (bf16)q0;
  Qb[obase + 2 * d2 + 1] = (bf16)q1;
  Kb[obase + 2 * d2] = (bf16)k0;
  Kb[obase + 2 * d2 + 1] = (bf16)k1;
  const size_t vtb = (size_t)(b * Hc + h) * HDc * Sc;
  Vt[vtb + (size_t)(2 * d2) * Sc + s] = (bf16)vp[2 * d2];
  Vt[vtb + (size_t)(2 * d2 + 1) * Sc + s] = (bf16)vp[2 * d2 + 1];
}

// ----------------------------------------------------- flash attention (WMMA)
// grid (S/64, B*H), 128 threads = 4 waves; wave owns 16 queries, full HD=256.
__global__ __launch_bounds__(128) void flash_kernel(const bf16* __restrict__ Qb,
                                                    const bf16* __restrict__ Kb,
                                                    const bf16* __restrict__ Vt,
                                                    bf16* __restrict__ attn) {
  __shared__ __align__(32) bf16 Plds[4][16 * 64];
  const int lane = threadIdx.x & 31;
  const int wv = threadIdx.x >> 5;
  const int ln = lane & 15, hl = lane >> 4;
  const int bh = blockIdx.y;
  const int b = bh / Hc, hh = bh % Hc;
  const int qbase = blockIdx.x * 64 + wv * 16;
  const bf16* Qg = Qb + (size_t)bh * Sc * HDc;
  const bf16* Kg = Kb + (size_t)bh * Sc * HDc;
  const bf16* Vg = Vt + (size_t)bh * HDc * Sc;

  const v8f vzero = {};
  v8f o[16];
#pragma unroll
  for (int i = 0; i < 16; i++) o[i] = vzero;
  float mrow[8], lrow[8];
#pragma unroll
  for (int i = 0; i < 8; i++) { mrow[i] = -1e30f; lrow[i] = 0.f; }
  const float scale = 0.0625f;  // 1/sqrt(256)
  const int nkb = blockIdx.x + 1;

  for (int kb = 0; kb < nkb; kb++) {
    const int k0 = kb * 64;
    v8f sacc[4];
#pragma unroll
    for (int nt = 0; nt < 4; nt++) sacc[nt] = vzero;
#pragma unroll
    for (int ks = 0; ks < 8; ks++) {
      const v16bf aq =
          *(const v16bf*)(Qg + (size_t)(qbase + ln) * HDc + ks * 32 + hl * 16);
#pragma unroll
      for (int nt = 0; nt < 4; nt++) {
        const v16bf bk =
            *(const v16bf*)(Kg + (size_t)(k0 + nt * 16 + ln) * HDc + ks * 32 + hl * 16);
        sacc[nt] = wmma_bf16(aq, bk, sacc[nt]);
      }
    }
    // scale + causal mask
#pragma unroll
    for (int nt = 0; nt < 4; nt++)
#pragma unroll
      for (int i = 0; i < 8; i++) {
        const int kj = k0 + nt * 16 + ln;
        const int qi = qbase + i + hl * 8;
        const float sv = sacc[nt][i] * scale;
        sacc[nt][i] = (kj > qi) ? -1e30f : sv;
      }
    // online softmax (row = VGPR index + half; reduce across 16 lanes of half)
    float mnew[8];
#pragma unroll
    for (int i = 0; i < 8; i++) {
      float rm = fmaxf(fmaxf(sacc[0][i], sacc[1][i]), fmaxf(sacc[2][i], sacc[3][i]));
#pragma unroll
      for (int off = 1; off < 16; off <<= 1) rm = fmaxf(rm, __shfl_xor(rm, off, 16));
      mnew[i] = fmaxf(mrow[i], rm);
    }
#pragma unroll
    for (int nt = 0; nt < 4; nt++)
#pragma unroll
      for (int i = 0; i < 8; i++) sacc[nt][i] = __expf(sacc[nt][i] - mnew[i]);
#pragma unroll
    for (int i = 0; i < 8; i++) {
      float r = sacc[0][i] + sacc[1][i] + sacc[2][i] + sacc[3][i];
#pragma unroll
      for (int off = 1; off < 16; off <<= 1) r += __shfl_xor(r, off, 16);
      const float alpha = __expf(mrow[i] - mnew[i]);
      lrow[i] = lrow[i] * alpha + r;
      mrow[i] = mnew[i];
#pragma unroll
      for (int j = 0; j < 16; j++) o[j][i] *= alpha;
    }
    // P (C-layout) -> LDS -> A-fragment layout
    bf16* P = &Plds[wv][0];
#pragma unroll
    for (int nt = 0; nt < 4; nt++)
#pragma unroll
      for (int i = 0; i < 8; i++)
        P[(i + hl * 8) * 64 + nt * 16 + ln] = (bf16)sacc[nt][i];
    asm volatile("s_wait_dscnt 0x0" ::: "memory");
#pragma unroll
    for (int ks = 0; ks < 2; ks++) {
      const v16bf pa = *(const v16bf*)(P + ln * 64 + ks * 32 + hl * 16);
#pragma unroll
      for (int nt = 0; nt < 16; nt++) {
        const v16bf bv =
            *(const v16bf*)(Vg + (size_t)(nt * 16 + ln) * Sc + k0 + ks * 32 + hl * 16);
        o[nt] = wmma_bf16(pa, bv, o[nt]);
      }
    }
  }
  // epilogue: normalize and scatter into [B*S][D] bf16 (col = h*HD + hd)
  float inv[8];
#pragma unroll
  for (int i = 0; i < 8; i++) inv[i] = 1.f / lrow[i];
#pragma unroll
  for (int nt = 0; nt < 16; nt++)
#pragma unroll
    for (int i = 0; i < 8; i++) {
      const int qi = qbase + i + hl * 8;
      const int col = hh * HDc + nt * 16 + ln;
      attn[(size_t)(b * Sc + qi) * Dc + col] = (bf16)(o[nt][i] * inv[i]);
    }
}

// --------------------------------------------------------------------------
extern "C" void kernel_launch(void* const* d_in, const int* in_sizes, int n_in,
                              void* d_out, int out_size, void* d_ws, size_t ws_size,
                              hipStream_t stream) {
  (void)in_sizes; (void)n_in; (void)out_size; (void)ws_size;
  const int* pos = (const int*)d_in[0];
  const float* hidden = (const float*)d_in[1];
  const float* ln_g = (const float*)d_in[2];
  const float* ln_b = (const float*)d_in[3];
  const float* w_qkv = (const float*)d_in[4];
  const float* w_out = (const float*)d_in[5];
  const float* w_fc_in = (const float*)d_in[6];
  const float* b_fc_in = (const float*)d_in[7];
  const float* w_fc_out = (const float*)d_in[8];
  const float* b_fc_out = (const float*)d_in[9];
  float* out = (float*)d_out;

  char* ws = (char*)d_ws;
  auto alloc = [&](size_t bytes) {
    char* p = ws;
    ws += (bytes + 255) & ~(size_t)255;
    return p;
  };
  bf16* h_bf  = (bf16*)alloc((size_t)MRc * Dc * 2);
  bf16* wqkvT = (bf16*)alloc((size_t)TDc * Dc * 2);
  bf16* woutT = (bf16*)alloc((size_t)Dc * Dc * 2);
  bf16* wfiT  = (bf16*)alloc((size_t)INNERc * Dc * 2);
  bf16* wfoT  = (bf16*)alloc((size_t)Dc * INNERc * 2);
  float* qkvf = (float*)alloc((size_t)MRc * TDc * 4);
  bf16* Qb    = (bf16*)alloc((size_t)Bc * Hc * Sc * HDc * 2);
  bf16* Kb    = (bf16*)alloc((size_t)Bc * Hc * Sc * HDc * 2);
  bf16* Vt    = (bf16*)alloc((size_t)Bc * Hc * Sc * HDc * 2);
  bf16* attnb = (bf16*)alloc((size_t)MRc * Dc * 2);
  bf16* mlpb  = (bf16*)alloc((size_t)MRc * INNERc * 2);

  ln_kernel<<<MRc, 256, 0, stream>>>(hidden, ln_g, ln_b, h_bf);

  transpose_cast<<<dim3(TDc / 32, Dc / 32), dim3(32, 8), 0, stream>>>(w_qkv, wqkvT, Dc, TDc);
  transpose_cast<<<dim3(Dc / 32, Dc / 32), dim3(32, 8), 0, stream>>>(w_out, woutT, Dc, Dc);
  transpose_cast<<<dim3(INNERc / 32, Dc / 32), dim3(32, 8), 0, stream>>>(w_fc_in, wfiT, Dc, INNERc);
  transpose_cast<<<dim3(Dc / 32, INNERc / 32), dim3(32, 8), 0, stream>>>(w_fc_out, wfoT, INNERc, Dc);

  // QKV projection (fp32 result for RoPE)
  gemm_wmma<0><<<dim3(TDc / 128, MRc / 128), 256, 0, stream>>>(
      h_bf, wqkvT, qkvf, nullptr, nullptr, nullptr, MRc, TDc, Dc);

  rope_kernel<<<Bc * Sc * Hc, 128, 0, stream>>>(qkvf, pos, Qb, Kb, Vt);

  flash_kernel<<<dim3(Sc / 64, Bc * Hc), 128, 0, stream>>>(Qb, Kb, Vt, attnb);

  // MLP: fc_in + gelu -> bf16
  gemm_wmma<3><<<dim3(INNERc / 128, MRc / 128), 256, 0, stream>>>(
      h_bf, wfiT, nullptr, mlpb, b_fc_in, nullptr, MRc, INNERc, Dc);
  // fc_out + bias + residual -> d_out (f32)
  gemm_wmma<1><<<dim3(Dc / 128, MRc / 128), 256, 0, stream>>>(
      mlpb, wfoT, out, nullptr, b_fc_out, hidden, MRc, Dc, INNERc);
  // attention out-projection accumulates into d_out
  gemm_wmma<2><<<dim3(Dc / 128, MRc / 128), 256, 0, stream>>>(
      attnb, woutT, out, nullptr, nullptr, nullptr, MRc, Dc, Dc);
}
